// GQAMultiheadAttention_17179869801
// MI455X (gfx1250) — matrix operand order repaired and checked by reading, not compiled, not run on hardware
//
#include <hip/hip_runtime.h>

// ---------------------------------------------------------------- types
typedef __attribute__((ext_vector_type(8)))  __bf16 v8bf;
typedef __attribute__((ext_vector_type(16))) __bf16 v16bf;
typedef __attribute__((ext_vector_type(8)))  float  v8f;

#define T_SEQ 2048
#define C_DIM 2048
#define KV_DIM 512
#define HD 64
#define N_HEADS 32
#define N_KV_HEADS 8

// ---------------------------------------------------------------- helpers
__device__ __forceinline__ unsigned short f2bfu(float f) {
    union { float f; unsigned int u; } cv; cv.f = f;
    unsigned int u = cv.u;
    return (unsigned short)((u + 0x7FFFu + ((u >> 16) & 1u)) >> 16);
}

__device__ __forceinline__ v16bf ldfrag(const unsigned short* lo, const unsigned short* hi) {
    v8bf a = *reinterpret_cast<const v8bf*>(lo);
    v8bf b = *reinterpret_cast<const v8bf*>(hi);
    return __builtin_shufflevector(a, b, 0,1,2,3,4,5,6,7,8,9,10,11,12,13,14,15);
}

__device__ __forceinline__ v8f wmma_bf16(v16bf a, v16bf b, v8f c) {
    // D = A(16x32 bf16) * B(32x16 bf16) + C(16x16 f32)
    return __builtin_amdgcn_wmma_f32_16x16x32_bf16(
        /*neg_a=*/false, a, /*neg_b=*/false, b,
        /*c_mod=*/(short)0, c, /*reuse_a=*/false, /*reuse_b=*/false);
}

// gfx1250 async copy global->LDS (ASYNCcnt); ldsOff = LDS byte address
// (low 32 bits of the generic pointer, per flat-aperture addressing).
__device__ __forceinline__ void async_ld_b128(unsigned ldsOff, const void* gaddr) {
    asm volatile("global_load_async_to_lds_b128 %0, %1, off"
                 :: "v"(ldsOff), "v"(gaddr) : "memory");
}
__device__ __forceinline__ void wait_async0() {
    asm volatile("s_wait_asynccnt 0x0" ::: "memory");
}

// ---------------------------------------------------------------- conversions
__global__ void cvt_bf16_kernel(const float* __restrict__ in,
                                unsigned short* __restrict__ out, size_t n) {
    size_t i = (size_t)blockIdx.x * blockDim.x + threadIdx.x;
    if (i < n) out[i] = f2bfu(in[i]);
}

// W[K][N] fp32 -> WT[N][K] bf16   (so columns of W become contiguous rows)
__global__ void cvt_wT_kernel(const float* __restrict__ W,
                              unsigned short* __restrict__ WT, int K, int N) {
    size_t i = (size_t)blockIdx.x * blockDim.x + threadIdx.x;
    if (i >= (size_t)K * N) return;
    int k = (int)(i % K);
    int n = (int)(i / K);
    WT[i] = f2bfu(W[(size_t)k * N + n]);
}

// RoPE on [T, nh, 64] fp32 -> bf16 (pairs (2i, 2i+1)); scale folds 1/sqrt(hd) for Q
__global__ void rope_kernel(const float* __restrict__ in,
                            const float* __restrict__ sn,
                            const float* __restrict__ cs,
                            unsigned short* __restrict__ out,
                            int nHeads, float scale) {
    size_t idx = (size_t)blockIdx.x * blockDim.x + threadIdx.x;
    size_t total = (size_t)T_SEQ * nHeads * (HD / 2);
    if (idx >= total) return;
    int i = (int)(idx % (HD / 2));
    int h = (int)((idx / (HD / 2)) % nHeads);
    int t = (int)(idx / ((size_t)(HD / 2) * nHeads));
    size_t base = (size_t)t * nHeads * HD + (size_t)h * HD + 2 * i;
    float x1 = in[base], x2 = in[base + 1];
    float s = sn[(size_t)t * 32 + i], c = cs[(size_t)t * 32 + i];
    out[base]     = f2bfu((x1 * c - x2 * s) * scale);
    out[base + 1] = f2bfu((x1 * s + x2 * c) * scale);
}

// V[T, 512] fp32 -> VT[kvh][d][T] bf16
__global__ void vtrans_kernel(const float* __restrict__ V,
                              unsigned short* __restrict__ VT) {
    size_t idx = (size_t)blockIdx.x * blockDim.x + threadIdx.x;
    if (idx >= (size_t)N_KV_HEADS * HD * T_SEQ) return;
    int t  = (int)(idx & (T_SEQ - 1));
    int d  = (int)((idx >> 11) & (HD - 1));
    int kh = (int)(idx >> 17);
    VT[idx] = f2bfu(V[(size_t)t * KV_DIM + kh * HD + d]);
}

// ---------------------------------------------------------------- GEMM
// C[M,N] f32 = A[M,K] bf16  x  BT[N,K] bf16  (B pre-transposed)
// Block: 256 thr = 8 waves; block tile 128x128; wave tile 32x64 (2x4 wmma tiles)
__global__ void __launch_bounds__(256)
gemm_bf16_kernel(const unsigned short* __restrict__ A,
                 const unsigned short* __restrict__ BT,
                 float* __restrict__ C, int M, int N, int K) {
    const int lane = threadIdx.x & 31;
    const int wave = threadIdx.x >> 5;
    const int wm = wave >> 1;          // 0..3
    const int wn = wave & 1;           // 0..1
    const int lm = lane & 15;
    const int h2 = lane >> 4;          // 0/1
    const int mBase = blockIdx.y * 128 + wm * 32;
    const int nBase = blockIdx.x * 128 + wn * 64;

    const unsigned short* aptr[2];
    const unsigned short* bptr[4];
    #pragma unroll
    for (int mt = 0; mt < 2; ++mt) aptr[mt] = A + (size_t)(mBase + mt * 16 + lm) * K;
    #pragma unroll
    for (int nt = 0; nt < 4; ++nt) bptr[nt] = BT + (size_t)(nBase + nt * 16 + lm) * K;

    v8f acc[2][4];
    #pragma unroll
    for (int mt = 0; mt < 2; ++mt)
        #pragma unroll
        for (int nt = 0; nt < 4; ++nt)
            #pragma unroll
            for (int j = 0; j < 8; ++j) acc[mt][nt][j] = 0.f;

    for (int k0 = 0; k0 < K; k0 += 32) {
        if (k0 + 32 < K) {  // prefetch next K-slab
            __builtin_prefetch(aptr[0] + k0 + 32, 0, 1);
            __builtin_prefetch(bptr[0] + k0 + 32, 0, 1);
        }
        v16bf af[2], bfg[4];
        #pragma unroll
        for (int mt = 0; mt < 2; ++mt)
            af[mt] = ldfrag(aptr[mt] + k0 + 8 * h2, aptr[mt] + k0 + 16 + 8 * h2);
        #pragma unroll
        for (int nt = 0; nt < 4; ++nt)
            bfg[nt] = ldfrag(bptr[nt] + k0 + 8 * h2, bptr[nt] + k0 + 16 + 8 * h2);
        #pragma unroll
        for (int mt = 0; mt < 2; ++mt)
            #pragma unroll
            for (int nt = 0; nt < 4; ++nt)
                acc[mt][nt] = wmma_bf16(af[mt], bfg[nt], acc[mt][nt]);
    }

    #pragma unroll
    for (int mt = 0; mt < 2; ++mt)
        #pragma unroll
        for (int nt = 0; nt < 4; ++nt)
            #pragma unroll
            for (int v = 0; v < 8; ++v) {
                int row = mBase + mt * 16 + v + (h2 << 3);
                int col = nBase + nt * 16 + lm;
                C[(size_t)row * N + col] = acc[mt][nt][v];
            }
}

// ---------------------------------------------------------------- SWA flash attention
// Grid (T/16, N_KV_HEADS); 128 threads = 4 waves, wave w serves head kh*4+w.
// The 4 heads of a GQA group share K/V: tiles are staged once per block into
// LDS with gfx1250 async copies (ASYNCcnt), double-buffered across kv blocks.
// S^T = K*Q^T (C-layout: lane = query) -> per-lane online softmax ->
// O^T = V^T*P^T -> transpose via LDS -> bf16 [T,C].
__global__ void __launch_bounds__(128)
attn_swa_kernel(const unsigned short* __restrict__ Qb,   // [T, C] bf16 (rope+scale)
                const unsigned short* __restrict__ Kb,   // [T, 512] bf16 (rope)
                const unsigned short* __restrict__ VT,   // [8][64][T] bf16
                const int* __restrict__ swa,
                unsigned short* __restrict__ AO) {       // [T, C] bf16
    const int q0   = blockIdx.x * 16;
    const int kh   = blockIdx.y;
    const int tid  = (int)threadIdx.x;
    const int wave = tid >> 5;
    const int lane = tid & 31;
    const int h    = kh * 4 + wave;        // query head for this wave
    const int lm   = lane & 15;
    const int h2   = lane >> 4;
    const int W    = *swa;

    __shared__ __align__(16) unsigned short sK[2][32 * 64];  // [kv][d]
    __shared__ __align__(16) unsigned short sV[2][64 * 32];  // [d][kv]
    __shared__ __align__(16) float sO[4][16 * 64];

    // Q^T B-fragments (K-dim = head dim, two 32-chunks), loop-invariant
    const unsigned short* qrow = Qb + (size_t)(q0 + lm) * C_DIM + h * HD;
    const v16bf bq0 = ldfrag(qrow + 8 * h2,      qrow + 16 + 8 * h2);
    const v16bf bq1 = ldfrag(qrow + 32 + 8 * h2, qrow + 48 + 8 * h2);

    float m_run = -1e30f, s_run = 0.f;
    v8f o[4];
    #pragma unroll
    for (int dt = 0; dt < 4; ++dt)
        #pragma unroll
        for (int j = 0; j < 8; ++j) o[dt][j] = 0.f;

    int kstart = q0 - W; if (kstart < 0) kstart = 0; kstart &= ~31;
    const int kend = (q0 + 15) & ~31;

    const unsigned short* kg = Kb + kh * HD;                   // + row*KV_DIM
    const unsigned short* vg = VT + (size_t)kh * HD * T_SEQ;   // + d*T + t
    const int q = q0 + lm;

    // cooperative async stage of one 32-key K/V tile into buffer b
    const int kr = tid >> 3, kc = (tid & 7) * 8;   // K: 8 lanes/row, 16 rows/issue
    const int vr = tid >> 2, vc = (tid & 3) * 8;   // V: 4 lanes/row, 32 rows/issue
    auto stage = [&](int b, int kvb) {
        #pragma unroll
        for (int i = 0; i < 2; ++i) {
            int r = kr + i * 16;
            async_ld_b128((unsigned)(uintptr_t)&sK[b][r * 64 + kc],
                          kg + (size_t)(kvb + r) * KV_DIM + kc);
        }
        #pragma unroll
        for (int i = 0; i < 2; ++i) {
            int d = vr + i * 32;
            async_ld_b128((unsigned)(uintptr_t)&sV[b][d * 32 + vc],
                          vg + (size_t)d * T_SEQ + kvb + vc);
        }
    };

    int buf = 0;
    stage(0, kstart);

    for (int kvb = kstart; kvb <= kend; kvb += 32) {
        wait_async0();        // this wave's staged tile is in LDS
        __syncthreads();      // everyone's quarter landed; prev compute done
        if (kvb + 32 <= kend) stage(buf ^ 1, kvb + 32);

        const unsigned short* kt = &sK[buf][0];
        const unsigned short* vt = &sV[buf][0];

        // ---- S^T tiles from LDS: tile0 = keys kvb..+15, tile1 = kvb+16..+31
        v8f c01[2];
        #pragma unroll
        for (int t = 0; t < 2; ++t) {
            const unsigned short* krow = kt + (t * 16 + lm) * 64;
            v16bf ak0 = ldfrag(krow + 8 * h2,      krow + 16 + 8 * h2);
            v16bf ak1 = ldfrag(krow + 32 + 8 * h2, krow + 48 + 8 * h2);
            v8f z;
            #pragma unroll
            for (int j = 0; j < 8; ++j) z[j] = 0.f;
            z = wmma_bf16(ak0, bq0, z);
            c01[t] = wmma_bf16(ak1, bq1, z);
        }

        // ---- mask + online softmax (per-lane query stats)
        float s0[8], s1[8], bm = -1e30f;
        #pragma unroll
        for (int v = 0; v < 8; ++v) {
            int kv0 = kvb + v + (h2 << 3);
            int r0 = q - kv0, r1 = q - (kv0 + 16);
            s0[v] = (r0 < 0 || r0 > W) ? -1e30f : c01[0][v];
            s1[v] = (r1 < 0 || r1 > W) ? -1e30f : c01[1][v];
            bm = fmaxf(bm, fmaxf(s0[v], s1[v]));
        }
        bm = fmaxf(bm, __shfl_xor(bm, 16));
        const float m_new = fmaxf(m_run, bm);
        const float scale = __expf(m_run - m_new);
        float p0[8], p1[8], bs = 0.f;
        #pragma unroll
        for (int v = 0; v < 8; ++v) {
            p0[v] = __expf(s0[v] - m_new);
            p1[v] = __expf(s1[v] - m_new);
            bs += p0[v] + p1[v];
        }
        bs += __shfl_xor(bs, 16);
        s_run = s_run * scale + bs;
        m_run = m_new;
        #pragma unroll
        for (int dt = 0; dt < 4; ++dt)
            #pragma unroll
            for (int j = 0; j < 8; ++j) o[dt][j] *= scale;

        // ---- repack P^T (C-layout) into B-operand layout (cross-half shuffle)
        float pp0[8], pp1[8];
        #pragma unroll
        for (int v = 0; v < 8; ++v) {
            pp0[v] = __shfl_xor(p0[v], 16);
            pp1[v] = __shfl_xor(p1[v], 16);
        }
        unsigned short e[16];
        #pragma unroll
        for (int v = 0; v < 8; ++v) {
            e[v]     = f2bfu(lane < 16 ? p0[v]  : pp1[v]);
            e[8 + v] = f2bfu(lane < 16 ? pp0[v] : p1[v]);
        }
        v16bf bp; __builtin_memcpy(&bp, e, 32);

        // ---- O^T += V^T * P^T  (4 d-tiles of 16) from LDS
        #pragma unroll
        for (int dt = 0; dt < 4; ++dt) {
            const unsigned short* vrow = vt + (dt * 16 + lm) * 32;
            v16bf av = ldfrag(vrow + 16 * h2, vrow + 16 * h2 + 8);
            o[dt] = wmma_bf16(av, bp, o[dt]);
        }
        buf ^= 1;
    }

    // ---- normalize, transpose via LDS, store bf16 [T, C]
    const float inv = 1.f / s_run;
    #pragma unroll
    for (int dt = 0; dt < 4; ++dt)
        #pragma unroll
        for (int v = 0; v < 8; ++v)
            sO[wave][lm * 64 + dt * 16 + v + (h2 << 3)] = o[dt][v] * inv;
    __syncthreads();

    const int rq = lane >> 1, dh = lane & 1;
    unsigned int pack[8];
    #pragma unroll
    for (int j = 0; j < 8; ++j) {
        float a = sO[wave][rq * 64 + dh * 32 + 2 * j];
        float b = sO[wave][rq * 64 + dh * 32 + 2 * j + 1];
        pack[j] = (unsigned int)f2bfu(a) | ((unsigned int)f2bfu(b) << 16);
    }
    unsigned short* dst = AO + (size_t)(q0 + rq) * C_DIM + h * HD + dh * 32;
    uint4 u0; u0.x = pack[0]; u0.y = pack[1]; u0.z = pack[2]; u0.w = pack[3];
    uint4 u1; u1.x = pack[4]; u1.y = pack[5]; u1.z = pack[6]; u1.w = pack[7];
    reinterpret_cast<uint4*>(dst)[0] = u0;
    reinterpret_cast<uint4*>(dst)[1] = u1;
}

// ---------------------------------------------------------------- launch
extern "C" void kernel_launch(void* const* d_in, const int* in_sizes, int n_in,
                              void* d_out, int out_size, void* d_ws, size_t ws_size,
                              hipStream_t stream) {
    const float* x   = (const float*)d_in[0];
    const float* Wq  = (const float*)d_in[1];
    const float* Wk  = (const float*)d_in[2];
    const float* Wv  = (const float*)d_in[3];
    const float* Wo  = (const float*)d_in[4];
    const float* rsn = (const float*)d_in[5];
    const float* rcs = (const float*)d_in[6];
    const int*   swa = (const int*)d_in[7];
    float* out = (float*)d_out;

    char* ws = (char*)d_ws;
    const size_t TC  = (size_t)T_SEQ * C_DIM;    // 4M elems
    const size_t TKV = (size_t)T_SEQ * KV_DIM;   // 1M elems
    size_t off = 0;
    unsigned short* xb  = (unsigned short*)(ws + off); off += TC * 2;
    unsigned short* WqT = (unsigned short*)(ws + off); off += TC * 2;
    unsigned short* WkT = (unsigned short*)(ws + off); off += TKV * 2;
    unsigned short* WvT = (unsigned short*)(ws + off); off += TKV * 2;
    unsigned short* WoT = (unsigned short*)(ws + off); off += TC * 2;
    float* Qf = (float*)(ws + off); off += TC * 4;
    float* Kf = (float*)(ws + off); off += TKV * 4;
    float* Vf = (float*)(ws + off); off += TKV * 4;
    unsigned short* Qb  = (unsigned short*)(ws + off); off += TC * 2;
    unsigned short* Kb  = (unsigned short*)(ws + off); off += TKV * 2;
    unsigned short* VTb = (unsigned short*)(ws + off); off += TKV * 2;
    unsigned short* AOb = (unsigned short*)(ws + off); off += TC * 2;
    (void)ws_size; (void)n_in; (void)in_sizes; (void)out_size;

    const int B = 256;
    // 1) bf16 conversions (+ weight transpose)
    cvt_bf16_kernel<<<(unsigned)((TC + B - 1) / B), B, 0, stream>>>(x, xb, TC);
    cvt_wT_kernel<<<(unsigned)((TC + B - 1) / B),  B, 0, stream>>>(Wq, WqT, C_DIM, C_DIM);
    cvt_wT_kernel<<<(unsigned)((TKV + B - 1) / B), B, 0, stream>>>(Wk, WkT, C_DIM, KV_DIM);
    cvt_wT_kernel<<<(unsigned)((TKV + B - 1) / B), B, 0, stream>>>(Wv, WvT, C_DIM, KV_DIM);
    cvt_wT_kernel<<<(unsigned)((TC + B - 1) / B),  B, 0, stream>>>(Wo, WoT, C_DIM, C_DIM);

    // 2) QKV projections (WMMA bf16)
    gemm_bf16_kernel<<<dim3(C_DIM / 128, T_SEQ / 128), 256, 0, stream>>>(
        xb, WqT, Qf, T_SEQ, C_DIM, C_DIM);
    gemm_bf16_kernel<<<dim3(KV_DIM / 128, T_SEQ / 128), 256, 0, stream>>>(
        xb, WkT, Kf, T_SEQ, KV_DIM, C_DIM);
    gemm_bf16_kernel<<<dim3(KV_DIM / 128, T_SEQ / 128), 256, 0, stream>>>(
        xb, WvT, Vf, T_SEQ, KV_DIM, C_DIM);

    // 3) RoPE (1/sqrt(64) folded into Q) + V transpose, all -> bf16
    const size_t nQ = (size_t)T_SEQ * N_HEADS * (HD / 2);
    const size_t nK = (size_t)T_SEQ * N_KV_HEADS * (HD / 2);
    rope_kernel<<<(unsigned)((nQ + B - 1) / B), B, 0, stream>>>(Qf, rsn, rcs, Qb, N_HEADS, 0.125f);
    rope_kernel<<<(unsigned)((nK + B - 1) / B), B, 0, stream>>>(Kf, rsn, rcs, Kb, N_KV_HEADS, 1.0f);
    vtrans_kernel<<<(unsigned)((TKV + B - 1) / B), B, 0, stream>>>(Vf, VTb);

    // 4) sliding-window flash attention (WMMA + async LDS staging, GQA-shared K/V)
    attn_swa_kernel<<<dim3(T_SEQ / 16, N_KV_HEADS), 128, 0, stream>>>(Qb, Kb, VTb, swa, AOb);

    // 5) output projection -> fp32 d_out
    gemm_bf16_kernel<<<dim3(C_DIM / 128, T_SEQ / 128), 256, 0, stream>>>(
        AOb, WoT, out, T_SEQ, C_DIM, C_DIM);
}
